// MultiQueryAttention_20993800143131
// MI455X (gfx1250) — matrix operand order repaired and checked by reading, not compiled, not run on hardware
//
#include <hip/hip_runtime.h>
#include <hip/hip_bf16.h>
#include <math.h>

#define D_MODEL 1024
#define NUM_HEADS 16
#define DIM_K 64
#define BATCH 2
#define SEQ 2048
#define ROWS (BATCH * SEQ)  // 4096

typedef __attribute__((ext_vector_type(16))) __bf16 v16bf;
typedef __attribute__((ext_vector_type(8)))  float  v8f;
typedef __attribute__((ext_vector_type(4)))  unsigned int v4u;
typedef __attribute__((ext_vector_type(4)))  int v4i;
typedef __attribute__((ext_vector_type(8)))  int v8i;

union Frag16 { uint4 u[2]; v16bf v; };

static __device__ __forceinline__ unsigned short f2bf(float f) {
  union { float f; unsigned u; } x; x.f = f;
  unsigned r = x.u + 0x7FFFu + ((x.u >> 16) & 1u);  // round-to-nearest-even
  return (unsigned short)(r >> 16);
}

// ---- Tensor Data Mover: async 2D tile load (global -> LDS), bf16 elements ----
// D# packing per CDNA5 ISA §8.3/8.4. Group0: count=1, lds_addr, global_addr[56:0],
// type=2. Group1: data_size=1 (2B), tensor_dim0/1, tile_dim0/1, tensor_dim0_stride.
// Groups 2/3 zero (tensor <= 2D). Issued by one wave; tracked by TENSORcnt.
static __device__ __forceinline__ void tdm_load_2d(unsigned lds_addr, const void* gptr,
                                                   unsigned dim0, unsigned dim1,
                                                   unsigned long long stride0,
                                                   unsigned tile0, unsigned tile1) {
  unsigned long long ga = (unsigned long long)(size_t)gptr;
  v4u g0;
  g0.x = 1u;                                                  // count=1, user D#
  g0.y = lds_addr;                                            // LDS byte address
  g0.z = (unsigned)(ga & 0xFFFFFFFFu);                        // global_addr[31:0]
  g0.w = (unsigned)((ga >> 32) & 0x01FFFFFFu) | (2u << 30);   // [56:32] | type=2
  v8i g1;
  g1[0] = (int)(1u << 16);                                    // data_size = 2 bytes
  g1[1] = (int)((dim0 & 0xFFFFu) << 16);                      // tensor_dim0[15:0]
  g1[2] = (int)((dim0 >> 16) | ((dim1 & 0xFFFFu) << 16));     // dim0[31:16]|dim1[15:0]
  g1[3] = (int)((dim1 >> 16) | (tile0 << 16));                // dim1[31:16]|tile_dim0
  g1[4] = (int)(tile1 & 0xFFFFu);                             // tile_dim1 (tile_dim2=0)
  g1[5] = (int)(stride0 & 0xFFFFFFFFull);                     // dim0_stride[31:0]
  g1[6] = (int)((stride0 >> 32) & 0xFFFFull);                 // dim0_stride[47:32]
  g1[7] = 0;
  v4i z4 = {0, 0, 0, 0};
#if defined(__clang_major__) && (__clang_major__ >= 23)
  v8i z8 = {0, 0, 0, 0, 0, 0, 0, 0};
  __builtin_amdgcn_tensor_load_to_lds(g0, g1, z4, z4, z8, 0);
#else
  __builtin_amdgcn_tensor_load_to_lds(g0, g1, z4, z4, 0);
#endif
}

// ---------------- fp32 -> bf16 elementwise ----------------
__global__ void cvt_bf16_kernel(const float* __restrict__ in,
                                unsigned short* __restrict__ out, int n) {
  int i = blockIdx.x * blockDim.x + threadIdx.x;
  if (i < n) out[i] = f2bf(in[i]);
}

// ------- transpose + convert: out[N,K] bf16 = in[K,N] fp32 -------
__global__ void transpose_cvt_kernel(const float* __restrict__ in,
                                     unsigned short* __restrict__ out,
                                     int K, int N) {
  int i = blockIdx.x * blockDim.x + threadIdx.x;
  if (i < K * N) {
    int n = i / K;
    int k = i - n * K;
    out[(size_t)n * K + k] = f2bf(in[(size_t)k * N + n]);
  }
}

// ---------------- bf16 WMMA GEMM: C[M,N] = A[M,K] @ WT[N,K]^T + bias ----------------
// Each wave computes a 16(M) x 64(N) tile; K-loop in steps of 32.
// mode 0: bf16 row-major [M,N]; mode 1: f32 row-major [M,N];
// mode 2: bf16 transposed "V^T" layout [BATCH][N][SEQ] (N must be 64).
__global__ void gemm_bf16_wmma_kernel(const unsigned short* __restrict__ A,
                                      const unsigned short* __restrict__ WT,
                                      const float* __restrict__ bias,
                                      void* __restrict__ Cout,
                                      int M, int N, int K, int mode) {
  int wid  = blockIdx.x * (blockDim.x >> 5) + (threadIdx.x >> 5);
  int lane = threadIdx.x & 31;
  int lo = lane & 15, hi = lane >> 4;
  int mTiles = M >> 4;
  int mTile  = wid % mTiles;
  int nGroup = wid / mTiles;
  int m = mTile * 16 + lo;

  v8f acc[4];
#pragma unroll
  for (int t = 0; t < 4; ++t) acc[t] = (v8f){0.f,0.f,0.f,0.f,0.f,0.f,0.f,0.f};

  const unsigned short* Arow = A + (size_t)m * K;
  for (int k0 = 0; k0 < K; k0 += 32) {
    __builtin_prefetch(Arow + k0 + 256, 0, 3);  // global_prefetch for next tiles
    Frag16 fa;  // A 16x32 bf16 fragment: lane row m, K = k0 + hi*8 (+16)
    fa.u[0] = *(const uint4*)(Arow + k0 + hi * 8);
    fa.u[1] = *(const uint4*)(Arow + k0 + 16 + hi * 8);
#pragma unroll
    for (int t = 0; t < 4; ++t) {
      int n = nGroup * 64 + t * 16 + lo;
      const unsigned short* Wrow = WT + (size_t)n * K + k0 + hi * 8;
      Frag16 fb;  // B 32x16 bf16 fragment: lane col n, K = k0 + hi*8 (+16)
      fb.u[0] = *(const uint4*)(Wrow);
      fb.u[1] = *(const uint4*)(Wrow + 16);
      acc[t] = __builtin_amdgcn_wmma_f32_16x16x32_bf16(
          false, fa.v, false, fb.v, (short)0, acc[t], false, false);
    }
  }

#pragma unroll
  for (int t = 0; t < 4; ++t) {
    int n = nGroup * 64 + t * 16 + lo;
    float bn = bias[n];
#pragma unroll
    for (int r = 0; r < 8; ++r) {
      int row = mTile * 16 + r + hi * 8;  // C layout: M = r + 8*hi, N = lo
      float val = acc[t][r] + bn;
      if (mode == 1) {
        ((float*)Cout)[(size_t)row * N + n] = val;
      } else if (mode == 0) {
        ((unsigned short*)Cout)[(size_t)row * N + n] = f2bf(val);
      } else {  // V^T: [b][n][s]
        int b = row >> 11, s = row & (SEQ - 1);
        ((unsigned short*)Cout)[((size_t)b * DIM_K + n) * SEQ + s] = f2bf(val);
      }
    }
  }
}

// ---------------- Flash MQA ----------------
// One workgroup = 16 waves = all 16 heads of one (batch, 16-row q tile).
// K/V chunks (shared across heads in MQA) are staged in LDS once per workgroup
// via the Tensor Data Mover; fragments then come from ds_load_b128.
__global__ void __launch_bounds__(512)
mqa_flash_kernel(const unsigned short* __restrict__ Qs,   // [ROWS,1024] bf16
                 const unsigned short* __restrict__ Ks,   // [ROWS,64]   bf16
                 const unsigned short* __restrict__ VT,   // [B*64,SEQ]  bf16
                 unsigned short* __restrict__ Out) {      // [ROWS,1024] bf16
  __shared__ __align__(16) unsigned short Kbuf[32][DIM_K];          // 4 KB: [key][kdim]
  __shared__ __align__(16) unsigned short Vbuf[DIM_K][32];          // 4 KB: [vdim][key]
  __shared__ __align__(16) unsigned short Plds[NUM_HEADS][16][32];  // 16 KB P staging

  int qTile = blockIdx.x & (SEQ / 16 - 1);  // 128 q tiles
  int b     = blockIdx.x >> 7;
  int h     = threadIdx.x >> 5;             // wave id == head
  int lane  = threadIdx.x & 31;
  int lo = lane & 15, hi = lane >> 4;
  int qBase = qTile * 16;

  // Q A-fragments for kdim [0,32) and [32,64): loaded once, reused all chunks
  const unsigned short* qrow =
      Qs + ((size_t)(b * SEQ + qBase + lo)) * D_MODEL + h * DIM_K;
  Frag16 qf[2];
#pragma unroll
  for (int kk = 0; kk < 2; ++kk) {
    qf[kk].u[0] = *(const uint4*)(qrow + kk * 32 + hi * 8);
    qf[kk].u[1] = *(const uint4*)(qrow + kk * 32 + 16 + hi * 8);
  }

  v8f o[4];
#pragma unroll
  for (int t = 0; t < 4; ++t) o[t] = (v8f){0.f,0.f,0.f,0.f,0.f,0.f,0.f,0.f};
  float mrow[8], lrow[8];
#pragma unroll
  for (int r = 0; r < 8; ++r) { mrow[r] = -1e30f; lrow[r] = 0.f; }

  int nKeys   = qBase + 16;             // causal: keys 0..qBase+15 (uniform per block)
  int nChunks = (nKeys + 31) >> 5;
  for (int c = 0; c < nChunks; ++c) {
    int kc = c * 32;
    __syncthreads();  // previous chunk's LDS tiles no longer in use
    if (h == 0) {
      // K chunk: 32 key rows x 64 kdim (contiguous rows, stride 64 elems)
      tdm_load_2d((unsigned)(size_t)&Kbuf[0][0],
                  Ks + ((size_t)(b * SEQ + kc)) * DIM_K,
                  DIM_K, 32, DIM_K, DIM_K, 32);
      // V^T chunk: 64 vdim rows x 32 keys (row stride SEQ elems)
      tdm_load_2d((unsigned)(size_t)&Vbuf[0][0],
                  VT + (size_t)b * DIM_K * SEQ + kc,
                  32, DIM_K, SEQ, 32, DIM_K);
      __builtin_amdgcn_s_wait_tensorcnt(0);
    }
    __syncthreads();  // K/V tiles visible to all waves

    v8f s01[2];
#pragma unroll
    for (int t = 0; t < 2; ++t) {
      int kt = kc + t * 16;
      v8f s = (v8f){0.f,0.f,0.f,0.f,0.f,0.f,0.f,0.f};
#pragma unroll
      for (int kk = 0; kk < 2; ++kk) {
        Frag16 kb;  // B frag: lane col = key t*16+lo, K = kdim kk*32 + hi*8 (+16)
        kb.u[0] = *(const uint4*)(&Kbuf[t * 16 + lo][kk * 32 + hi * 8]);
        kb.u[1] = *(const uint4*)(&Kbuf[t * 16 + lo][kk * 32 + 16 + hi * 8]);
        s = __builtin_amdgcn_wmma_f32_16x16x32_bf16(
            false, qf[kk].v, false, kb.v, (short)0, s, false, false);
      }
      // scale by 1/sqrt(64) and causal mask
#pragma unroll
      for (int r = 0; r < 8; ++r) {
        int rowSeq = qBase + r + hi * 8;
        int col    = kt + lo;
        s[r] = (col <= rowSeq) ? s[r] * 0.125f : -1e30f;
      }
      s01[t] = s;
    }

    // online softmax: row stats live in the 16-lane half that owns M = r + 8*hi
#pragma unroll
    for (int r = 0; r < 8; ++r) {
      float mx = fmaxf(s01[0][r], s01[1][r]);
#pragma unroll
      for (int off = 1; off < 16; off <<= 1) mx = fmaxf(mx, __shfl_xor(mx, off, 32));
      float mnew  = fmaxf(mrow[r], mx);
      float alpha = __expf(mrow[r] - mnew);
      mrow[r] = mnew;
      float p0 = __expf(s01[0][r] - mnew);
      float p1 = __expf(s01[1][r] - mnew);
      float rs = p0 + p1;
#pragma unroll
      for (int off = 1; off < 16; off <<= 1) rs += __shfl_xor(rs, off, 32);
      lrow[r] = lrow[r] * alpha + rs;
#pragma unroll
      for (int t = 0; t < 4; ++t) o[t][r] *= alpha;
      Plds[h][r + hi * 8][lo]      = f2bf(p0);
      Plds[h][r + hi * 8][16 + lo] = f2bf(p1);
    }
    __syncthreads();  // P tiles written (also orders within-wave LDS use)
    Frag16 pa;        // re-read P in A-fragment layout
    pa.u[0] = *(const uint4*)(&Plds[h][lo][hi * 8]);
    pa.u[1] = *(const uint4*)(&Plds[h][lo][16 + hi * 8]);

    // O += P(16x32) @ V(32x64): B frags contiguous along keys via V^T tile
#pragma unroll
    for (int t = 0; t < 4; ++t) {
      Frag16 vb;
      vb.u[0] = *(const uint4*)(&Vbuf[t * 16 + lo][hi * 8]);
      vb.u[1] = *(const uint4*)(&Vbuf[t * 16 + lo][16 + hi * 8]);
      o[t] = __builtin_amdgcn_wmma_f32_16x16x32_bf16(
          false, pa.v, false, vb.v, (short)0, o[t], false, false);
    }
  }

  // normalize and store bf16 attention output (heads concatenated)
#pragma unroll
  for (int t = 0; t < 4; ++t) {
#pragma unroll
    for (int r = 0; r < 8; ++r) {
      int rowSeq = qBase + r + hi * 8;
      float val = o[t][r] / lrow[r];
      Out[((size_t)(b * SEQ + rowSeq)) * D_MODEL + h * DIM_K + t * 16 + lo] = f2bf(val);
    }
  }
}

extern "C" void kernel_launch(void* const* d_in, const int* in_sizes, int n_in,
                              void* d_out, int out_size, void* d_ws, size_t ws_size,
                              hipStream_t stream) {
  const float* inq = (const float*)d_in[0];
  const float* ink = (const float*)d_in[1];
  const float* inv = (const float*)d_in[2];
  const float* Wq  = (const float*)d_in[3];
  const float* bq  = (const float*)d_in[4];
  const float* Wk  = (const float*)d_in[5];
  const float* bk  = (const float*)d_in[6];
  const float* Wv  = (const float*)d_in[7];
  const float* bv  = (const float*)d_in[8];
  const float* Wo  = (const float*)d_in[9];
  const float* bo  = (const float*)d_in[10];

  // workspace layout (bytes), total ~45.3 MB
  char* ws = (char*)d_ws;
  unsigned short* Xq   = (unsigned short*)(ws + 0);          // 8 MB  [4096,1024]
  unsigned short* Xk   = (unsigned short*)(ws + 8388608);    // 8 MB
  unsigned short* Xv   = (unsigned short*)(ws + 16777216);   // 8 MB
  unsigned short* WqT  = (unsigned short*)(ws + 25165824);   // 2 MB  [1024,1024]
  unsigned short* WkT  = (unsigned short*)(ws + 27262976);   // 128 KB [64,1024]
  unsigned short* WvT  = (unsigned short*)(ws + 27394048);   // 128 KB
  unsigned short* WoT  = (unsigned short*)(ws + 27525120);   // 2 MB
  unsigned short* Qs   = (unsigned short*)(ws + 29622272);   // 8 MB  [4096,1024]
  unsigned short* Ks   = (unsigned short*)(ws + 38010880);   // 512 KB [4096,64]
  unsigned short* VT   = (unsigned short*)(ws + 38535168);   // 512 KB [B*64,2048]
  unsigned short* AtO  = (unsigned short*)(ws + 39059456);   // 8 MB  [4096,1024]

  const int nIn = ROWS * D_MODEL;  // 4 M elements
  cvt_bf16_kernel<<<nIn / 256, 256, 0, stream>>>(inq, Xq, nIn);
  cvt_bf16_kernel<<<nIn / 256, 256, 0, stream>>>(ink, Xk, nIn);
  cvt_bf16_kernel<<<nIn / 256, 256, 0, stream>>>(inv, Xv, nIn);

  transpose_cvt_kernel<<<(1024 * 1024) / 256, 256, 0, stream>>>(Wq, WqT, 1024, 1024);
  transpose_cvt_kernel<<<(1024 * 64) / 256, 256, 0, stream>>>(Wk, WkT, 1024, 64);
  transpose_cvt_kernel<<<(1024 * 64) / 256, 256, 0, stream>>>(Wv, WvT, 1024, 64);
  transpose_cvt_kernel<<<(1024 * 1024) / 256, 256, 0, stream>>>(Wo, WoT, 1024, 1024);

  // Q projection: [4096,1024] -> bf16; waves = 256 mTiles * 16 nGroups = 4096 -> 512 blocks
  gemm_bf16_wmma_kernel<<<512, 256, 0, stream>>>(Xq, WqT, bq, Qs,
                                                 ROWS, D_MODEL, D_MODEL, 0);
  // K projection: [4096,64] -> bf16; waves = 256 -> 32 blocks
  gemm_bf16_wmma_kernel<<<32, 256, 0, stream>>>(Xk, WkT, bk, Ks,
                                                ROWS, DIM_K, D_MODEL, 0);
  // V projection -> transposed [B][64][S]
  gemm_bf16_wmma_kernel<<<32, 256, 0, stream>>>(Xv, WvT, bv, VT,
                                                ROWS, DIM_K, D_MODEL, 2);
  // flash attention: one 16-wave block per (b, qTile); 256 blocks
  mqa_flash_kernel<<<BATCH * (SEQ / 16), 512, 0, stream>>>(Qs, Ks, VT, AtO);
  // output projection -> fp32 d_out
  gemm_bf16_wmma_kernel<<<512, 256, 0, stream>>>(AtO, WoT, bo, (float*)d_out,
                                                 ROWS, D_MODEL, D_MODEL, 1);
  (void)in_sizes; (void)n_in; (void)out_size; (void)ws_size;
}
